// MegaBlocksMoeMLP_44341242364382
// MI455X (gfx1250) — matrix operand order repaired and checked by reading, not compiled, not run on hardware
//
#include <hip/hip_runtime.h>
#include <hip/hip_bf16.h>
#include <stdint.h>

#define NEXP 8
#define T_TOK 2048
#define H_DIM 1024
#define I_DIM 1024
#define N1 (2 * I_DIM)
#define ALPHA_C 1.702f
#define LIMIT_C 7.0f

#define BM 128      // block M tile
#define BNB 128     // block N tile
#define BWN 64      // per-wave N span
#define BK 32
#define NIT1 (H_DIM / BK)
#define NIT2 (I_DIM / BK)

typedef __attribute__((ext_vector_type(16))) __bf16 v16bf;
typedef __attribute__((ext_vector_type(8)))  float  v8f;

union AB16 { uint32_t u[8]; v16bf v; };

__device__ __forceinline__ unsigned short f32_to_bf16(float f) {
  uint32_t u = __builtin_bit_cast(uint32_t, f);
  uint32_t r = u + 0x7FFFu + ((u >> 16) & 1u);  // round-to-nearest-even
  return (unsigned short)(r >> 16);
}

#if __has_builtin(__builtin_amdgcn_cvt_pk_bf16_f32)
typedef __attribute__((ext_vector_type(2))) __bf16 v2bf;
__device__ __forceinline__ uint32_t pack2_bf16(float a, float b) {
  v2bf t = __builtin_amdgcn_cvt_pk_bf16_f32(a, b);
  return __builtin_bit_cast(uint32_t, t);
}
#else
__device__ __forceinline__ uint32_t pack2_bf16(float a, float b) {
  return (uint32_t)f32_to_bf16(a) | ((uint32_t)f32_to_bf16(b) << 16);
}
#endif

// ---------------- prep: zero output region, pack x -> bf16 pairs ----------------
__global__ void prep_kernel(const float* __restrict__ x, float* __restrict__ out,
                            uint32_t* __restrict__ xb) {
  int i = blockIdx.x * blockDim.x + threadIdx.x;  // over T*H/4
  if (i < T_TOK * H_DIM / 4) {
    float4 v = ((const float4*)x)[i];
    uint2 p;
    p.x = pack2_bf16(v.x, v.y);
    p.y = pack2_bf16(v.z, v.w);
    ((uint2*)xb)[i] = p;
    float4 z; z.x = 0.f; z.y = 0.f; z.z = 0.f; z.w = 0.f;
    ((float4*)out)[i] = z;
  }
}

// ---------------- router: one wave per token ----------------
__global__ void router_kernel(const float* __restrict__ x, const float* __restrict__ rw,
                              const float* __restrict__ rb, float* __restrict__ ew_out,
                              int* __restrict__ tkid, float* __restrict__ tkw) {
  int wave = threadIdx.x >> 5;
  int lane = threadIdx.x & 31;
  int t = blockIdx.x * 8 + wave;
  if (t >= T_TOK) return;
  float acc[NEXP];
#pragma unroll
  for (int e = 0; e < NEXP; e++) acc[e] = 0.f;
  const float* xr = x + (size_t)t * H_DIM;
  for (int k = lane; k < H_DIM; k += 32) {
    float xv = xr[k];
#pragma unroll
    for (int e = 0; e < NEXP; e++) acc[e] += xv * rw[e * H_DIM + k];
  }
#pragma unroll
  for (int e = 0; e < NEXP; e++) {
#pragma unroll
    for (int off = 16; off > 0; off >>= 1) acc[e] += __shfl_xor(acc[e], off, 32);
  }
  if (lane == 0) {
    float lg[NEXP];
    float mx = -1e30f;
#pragma unroll
    for (int e = 0; e < NEXP; e++) { lg[e] = acc[e] + rb[e]; mx = fmaxf(mx, lg[e]); }
    float s = 0.f;
#pragma unroll
    for (int e = 0; e < NEXP; e++) { lg[e] = __expf(lg[e] - mx); s += lg[e]; }
    float inv = 1.f / s;
#pragma unroll
    for (int e = 0; e < NEXP; e++) lg[e] *= inv;
    int b0 = 0; float p0 = lg[0];
#pragma unroll
    for (int e = 1; e < NEXP; e++) if (lg[e] > p0) { p0 = lg[e]; b0 = e; }
    int b1 = (b0 == 0) ? 1 : 0; float p1 = lg[b1];
#pragma unroll
    for (int e = 0; e < NEXP; e++) if (e != b0 && lg[e] > p1) { p1 = lg[e]; b1 = e; }
    ew_out[t * 2 + 0] = p0; ew_out[t * 2 + 1] = p1;
    tkid[t * 2 + 0] = b0;  tkid[t * 2 + 1] = b1;
    tkw[t * 2 + 0] = p0;   tkw[t * 2 + 1] = p1;
  }
}

// ---------------- build per-expert slot lists (deterministic, token order) ----------------
__global__ void build_lists_kernel(const int* __restrict__ tkid, const float* __restrict__ tkw,
                                   int* __restrict__ counts, int* __restrict__ offs,
                                   int* __restrict__ slot_tok, float* __restrict__ slot_w) {
  __shared__ int scnt[NEXP];
  __shared__ int soff[NEXP];
  int e = threadIdx.x >> 5;
  int lane = threadIdx.x & 31;
  int cnt = 0;
  for (int t0 = 0; t0 < T_TOK; t0 += 32) {
    int t = t0 + lane;
    bool f = (tkid[t * 2] == e) || (tkid[t * 2 + 1] == e);
    unsigned long long m = __ballot(f);
    cnt += __popcll(m);
  }
  if (lane == 0) scnt[e] = cnt;
  __syncthreads();
  if (threadIdx.x == 0) {
    int o = 0;
    for (int i = 0; i < NEXP; i++) { soff[i] = o; o += scnt[i]; }
  }
  __syncthreads();
  int pos = soff[e];
  for (int t0 = 0; t0 < T_TOK; t0 += 32) {
    int t = t0 + lane;
    int i0 = tkid[t * 2], i1 = tkid[t * 2 + 1];
    bool f = (i0 == e) || (i1 == e);
    unsigned long long m = __ballot(f);
    int pre = __popcll(m & ((1ull << lane) - 1ull));
    if (f) {
      slot_tok[pos + pre] = t;
      slot_w[pos + pre] = (i0 == e) ? tkw[t * 2] : tkw[t * 2 + 1];
    }
    pos += __popcll(m);
  }
  if (lane == 0) { counts[e] = scnt[e]; offs[e] = soff[e]; }
}

// ======================= GEMM core helpers =======================
// LDS tiles: lA [m][k] (row = 32 bf16), lB transposed [n][k].
// Fragment packing per ISA 7.12.2 (16-bit A 16x32, B 32x16), wave32.

__device__ __forceinline__ void load_a_frag(const unsigned short* lA, int arow, int hi,
                                            AB16& Af) {
  const uint32_t* aU = (const uint32_t*)(lA + arow * BK);
  int kb = hi * 4;
#pragma unroll
  for (int j = 0; j < 4; j++) Af.u[j] = aU[kb + j];
#pragma unroll
  for (int j = 0; j < 4; j++) Af.u[4 + j] = aU[kb + 8 + j];
}

__device__ __forceinline__ void load_b_frag(const unsigned short* lB, int ncol, int hi,
                                            AB16& Bf) {
  const uint32_t* bU = (const uint32_t*)(lB + ncol * BK);
  int kb = hi * 4;
#pragma unroll
  for (int j = 0; j < 8; j++) Bf.u[j] = bU[kb + (j & 3) + (j >> 2) * 8];
}

// ---------------- GEMM1: h = gather(xb) @ w1[e]; fused bias + clamped GLU -> act(bf16) ----
__global__ __launch_bounds__(256) void gemm1_kernel(
    const uint32_t* __restrict__ xb, const float* __restrict__ w1,
    const float* __restrict__ w1b, const int* __restrict__ counts,
    const int* __restrict__ offs, const int* __restrict__ slot_tok,
    unsigned short* __restrict__ act) {
  int e = blockIdx.z;
  int cnt = counts[e];
  int m0 = blockIdx.y * BM;
  if (m0 >= cnt) return;
  int n0 = blockIdx.x * BNB;
  int aoff = offs[e];

  __shared__ __align__(16) unsigned short lA[2][BM * BK];
  __shared__ __align__(16) unsigned short lB[2][BNB * BK];

  int tid = threadIdx.x;
  int lane = tid & 31;
  int w = tid >> 5;         // 8 waves
  int wm = w & 3;           // rows [32*wm, 32*wm+32)
  int wn = w >> 2;          // cols [64*wn, 64*wn+64)
  int hi = lane >> 4;
  int ml = lane & 15;

  // A staging: two threads per row (halves)
  int arowi = tid >> 1;
  int ahalf = tid & 1;
  int mrow = m0 + arowi;
  int mytok = slot_tok[aoff + ((mrow < cnt) ? mrow : 0)];
  const uint4* xrow = (const uint4*)xb + (size_t)mytok * (H_DIM / 8) + ahalf * 2;
  unsigned short* aSt0 = lA[0] + arowi * BK + ahalf * 16;
  unsigned short* aSt1 = lA[1] + arowi * BK + ahalf * 16;

  // B staging: thread covers n = tid&127, k = (tid>>7)*16 + i (k-adjacent pairs packable)
  int bn = tid & 127;
  int bks = (tid >> 7) * 16;
  const float* wbase = w1 + (size_t)e * H_DIM * N1 + (size_t)bks * N1 + n0 + bn;
  uint32_t* bSt0 = (uint32_t*)(lB[0] + bn * BK + bks);
  uint32_t* bSt1 = (uint32_t*)(lB[1] + bn * BK + bks);

  v8f acc[2][4];
#pragma unroll
  for (int ma = 0; ma < 2; ma++)
#pragma unroll
    for (int f = 0; f < 4; f++)
#pragma unroll
      for (int r = 0; r < 8; r++) acc[ma][f][r] = 0.f;

  uint4 ra[2];
  float rbv[16];

  // prologue: tile 0
#pragma unroll
  for (int q = 0; q < 2; q++) ra[q] = xrow[q];
#pragma unroll
  for (int i = 0; i < 16; i++) rbv[i] = wbase[(size_t)i * N1];
#pragma unroll
  for (int q = 0; q < 2; q++) ((uint4*)aSt0)[q] = ra[q];
#pragma unroll
  for (int p = 0; p < 8; p++) bSt0[p] = pack2_bf16(rbv[2 * p], rbv[2 * p + 1]);
  __syncthreads();

  for (int it = 0; it < NIT1; ++it) {
    int cur = it & 1;
    if (it + 1 < NIT1) {
      int k0 = (it + 1) * BK;
#pragma unroll
      for (int q = 0; q < 2; q++) ra[q] = xrow[(k0 >> 3) + q];
      const float* wb = wbase + (size_t)k0 * N1;
#pragma unroll
      for (int i = 0; i < 16; i++) rbv[i] = wb[(size_t)i * N1];
    }
    AB16 Af0, Af1;
    load_a_frag(lA[cur], 32 * wm + ml, hi, Af0);
    load_a_frag(lA[cur], 32 * wm + 16 + ml, hi, Af1);
#pragma unroll
    for (int f = 0; f < 4; f++) {
      AB16 Bf;
      load_b_frag(lB[cur], 64 * wn + 16 * f + ml, hi, Bf);
      acc[0][f] = __builtin_amdgcn_wmma_f32_16x16x32_bf16(
          false, Af0.v, false, Bf.v, (short)0, acc[0][f], false, false);
      acc[1][f] = __builtin_amdgcn_wmma_f32_16x16x32_bf16(
          false, Af1.v, false, Bf.v, (short)0, acc[1][f], false, false);
    }
    if (it + 1 < NIT1) {
      unsigned short* aDst = (cur ? aSt0 : aSt1);
      uint32_t* bDst = (cur ? bSt0 : bSt1);
#pragma unroll
      for (int q = 0; q < 2; q++) ((uint4*)aDst)[q] = ra[q];
#pragma unroll
      for (int p = 0; p < 8; p++) bDst[p] = pack2_bf16(rbv[2 * p], rbv[2 * p + 1]);
    }
    __syncthreads();
  }

  // epilogue: gate/up pairing via lane shuffle (cols 2j,2j+1 are adjacent lanes)
  int mlq = ml >> 1;
  bool evenlane = (ml & 1) == 0;
#pragma unroll
  for (int ma = 0; ma < 2; ma++) {
#pragma unroll
    for (int f = 0; f < 4; f++) {
      int n = n0 + 64 * wn + 16 * f + ml;
      int nb = n & ~1;
      float bg = w1b[e * N1 + nb];
      float bu = w1b[e * N1 + nb + 1];
#pragma unroll
      for (int r = 0; r < 8; r++) {
        float a = acc[ma][f][r];
        float p = __shfl_xor(a, 1, 32);
        if (evenlane) {
          float gate = fminf(a + bg, LIMIT_C);
          float up = fminf(fmaxf(p + bu, -LIMIT_C), LIMIT_C);
          float glu = gate / (1.0f + __expf(-ALPHA_C * gate));
          float av = (up + 1.0f) * glu;
          int m = m0 + 32 * wm + 16 * ma + hi * 8 + r;
          if (m < cnt) {
            act[(size_t)(aoff + m) * I_DIM + (n0 >> 1) + 32 * wn + 8 * f + mlq] =
                f32_to_bf16(av);
          }
        }
      }
    }
  }
}

// ---------------- GEMM2: y = act @ w2[e] + b2, combine-scaled atomic accumulate ----------
__global__ __launch_bounds__(256) void gemm2_kernel(
    const unsigned short* __restrict__ act, const float* __restrict__ w2,
    const float* __restrict__ w2b, const int* __restrict__ counts,
    const int* __restrict__ offs, const int* __restrict__ slot_tok,
    const float* __restrict__ slot_w, float* __restrict__ out) {
  int e = blockIdx.z;
  int cnt = counts[e];
  int m0 = blockIdx.y * BM;
  if (m0 >= cnt) return;
  int n0 = blockIdx.x * BNB;
  int aoff = offs[e];

  __shared__ __align__(16) unsigned short lA[2][BM * BK];
  __shared__ __align__(16) unsigned short lB[2][BNB * BK];

  int tid = threadIdx.x;
  int lane = tid & 31;
  int w = tid >> 5;
  int wm = w & 3;
  int wn = w >> 2;
  int hi = lane >> 4;
  int ml = lane & 15;

  int arowi = tid >> 1;
  int ahalf = tid & 1;
  int mrow = m0 + arowi;
  int myslot = aoff + ((mrow < cnt) ? mrow : 0);
  const uint4* arow = (const uint4*)act + (size_t)myslot * (I_DIM / 8) + ahalf * 2;
  unsigned short* aSt0 = lA[0] + arowi * BK + ahalf * 16;
  unsigned short* aSt1 = lA[1] + arowi * BK + ahalf * 16;

  int bn = tid & 127;
  int bks = (tid >> 7) * 16;
  const float* wbase = w2 + (size_t)e * I_DIM * H_DIM + (size_t)bks * H_DIM + n0 + bn;
  uint32_t* bSt0 = (uint32_t*)(lB[0] + bn * BK + bks);
  uint32_t* bSt1 = (uint32_t*)(lB[1] + bn * BK + bks);

  v8f acc[2][4];
#pragma unroll
  for (int ma = 0; ma < 2; ma++)
#pragma unroll
    for (int f = 0; f < 4; f++)
#pragma unroll
      for (int r = 0; r < 8; r++) acc[ma][f][r] = 0.f;

  uint4 ra[2];
  float rbv[16];

#pragma unroll
  for (int q = 0; q < 2; q++) ra[q] = arow[q];
#pragma unroll
  for (int i = 0; i < 16; i++) rbv[i] = wbase[(size_t)i * H_DIM];
#pragma unroll
  for (int q = 0; q < 2; q++) ((uint4*)aSt0)[q] = ra[q];
#pragma unroll
  for (int p = 0; p < 8; p++) bSt0[p] = pack2_bf16(rbv[2 * p], rbv[2 * p + 1]);
  __syncthreads();

  for (int it = 0; it < NIT2; ++it) {
    int cur = it & 1;
    if (it + 1 < NIT2) {
      int k0 = (it + 1) * BK;
#pragma unroll
      for (int q = 0; q < 2; q++) ra[q] = arow[(k0 >> 3) + q];
      const float* wb = wbase + (size_t)k0 * H_DIM;
#pragma unroll
      for (int i = 0; i < 16; i++) rbv[i] = wb[(size_t)i * H_DIM];
    }
    AB16 Af0, Af1;
    load_a_frag(lA[cur], 32 * wm + ml, hi, Af0);
    load_a_frag(lA[cur], 32 * wm + 16 + ml, hi, Af1);
#pragma unroll
    for (int f = 0; f < 4; f++) {
      AB16 Bf;
      load_b_frag(lB[cur], 64 * wn + 16 * f + ml, hi, Bf);
      acc[0][f] = __builtin_amdgcn_wmma_f32_16x16x32_bf16(
          false, Af0.v, false, Bf.v, (short)0, acc[0][f], false, false);
      acc[1][f] = __builtin_amdgcn_wmma_f32_16x16x32_bf16(
          false, Af1.v, false, Bf.v, (short)0, acc[1][f], false, false);
    }
    if (it + 1 < NIT2) {
      unsigned short* aDst = (cur ? aSt0 : aSt1);
      uint32_t* bDst = (cur ? bSt0 : bSt1);
#pragma unroll
      for (int q = 0; q < 2; q++) ((uint4*)aDst)[q] = ra[q];
#pragma unroll
      for (int p = 0; p < 8; p++) bDst[p] = pack2_bf16(rbv[2 * p], rbv[2 * p + 1]);
    }
    __syncthreads();
  }

  // epilogue: bias, combine-weight scale, atomic accumulate (exactly 2 adds/out elem)
  int toks[2][8]; float cws[2][8]; bool val[2][8];
#pragma unroll
  for (int ma = 0; ma < 2; ma++) {
#pragma unroll
    for (int r = 0; r < 8; r++) {
      int m = m0 + 32 * wm + 16 * ma + hi * 8 + r;
      val[ma][r] = (m < cnt);
      int slot = aoff + (val[ma][r] ? m : 0);
      toks[ma][r] = slot_tok[slot];
      cws[ma][r] = slot_w[slot];
    }
  }
#pragma unroll
  for (int f = 0; f < 4; f++) {
    int n = n0 + 64 * wn + 16 * f + ml;
    float bias = w2b[e * H_DIM + n];
#pragma unroll
    for (int ma = 0; ma < 2; ma++) {
#pragma unroll
      for (int r = 0; r < 8; r++) {
        if (val[ma][r]) {
          atomicAdd(out + (size_t)toks[ma][r] * H_DIM + n,
                    cws[ma][r] * (acc[ma][f][r] + bias));
        }
      }
    }
  }
}

extern "C" void kernel_launch(void* const* d_in, const int* in_sizes, int n_in,
                              void* d_out, int out_size, void* d_ws, size_t ws_size,
                              hipStream_t stream) {
  (void)in_sizes; (void)n_in; (void)out_size; (void)ws_size;
  const float* x   = (const float*)d_in[0];
  const float* rw  = (const float*)d_in[1];
  const float* rb  = (const float*)d_in[2];
  const float* w1  = (const float*)d_in[3];
  const float* w1b = (const float*)d_in[4];
  const float* w2  = (const float*)d_in[5];
  const float* w2b = (const float*)d_in[6];

  float* out = (float*)d_out;                       // (B,S,H) = T*H floats
  float* ew  = out + (size_t)T_TOK * H_DIM;         // (T, 2) expert weights

  char* ws = (char*)d_ws;
  int* counts = (int*)ws;            ws += NEXP * sizeof(int);
  int* offs   = (int*)ws;            ws += NEXP * sizeof(int);
  int* tkid   = (int*)ws;            ws += (size_t)T_TOK * 2 * sizeof(int);
  float* tkw  = (float*)ws;          ws += (size_t)T_TOK * 2 * sizeof(float);
  int* slot_tok = (int*)ws;          ws += (size_t)2 * T_TOK * sizeof(int);
  float* slot_w = (float*)ws;        ws += (size_t)2 * T_TOK * sizeof(float);
  ws = (char*)(((uintptr_t)ws + 127) & ~(uintptr_t)127);
  uint32_t* xb = (uint32_t*)ws;      ws += (size_t)T_TOK * H_DIM / 2 * sizeof(uint32_t);
  unsigned short* act = (unsigned short*)ws;  // 2T x I bf16

  prep_kernel<<<(T_TOK * H_DIM / 4 + 255) / 256, 256, 0, stream>>>(x, out, xb);
  router_kernel<<<T_TOK / 8, 256, 0, stream>>>(x, rw, rb, ew, tkid, tkw);
  build_lists_kernel<<<1, 256, 0, stream>>>(tkid, tkw, counts, offs, slot_tok, slot_w);
  gemm1_kernel<<<dim3(N1 / BNB, T_TOK / BM, NEXP), 256, 0, stream>>>(
      xb, w1, w1b, counts, offs, slot_tok, act);
  gemm2_kernel<<<dim3(H_DIM / BNB, T_TOK / BM, NEXP), 256, 0, stream>>>(
      act, w2, w2b, counts, offs, slot_tok, slot_w, out);
}
